// ProjectedAdaptiveLogSoftmax_18270790877731
// MI455X (gfx1250) — matrix operand order repaired and checked by reading, compiled, not run on hardware
//
#include <hip/hip_runtime.h>
#include <hip/hip_bf16.h>
#include <cstdint>

typedef __bf16 bf16_t;
typedef __bf16 v16bf __attribute__((ext_vector_type(16)));
typedef __bf16 v8bf  __attribute__((ext_vector_type(8)));
typedef float  v8f   __attribute__((ext_vector_type(8)));

#define NTOK   2048
#define DIM    1024
#define VHEAD  20002
#define VHEADP 20096   /* 157*128 */
#define V1     30000
#define V1P    30080   /* 235*128 */
#define V2     50000
#define V2P    50048   /* 391*128 */
#define TNH    157
#define TN1    235
#define TN2    391

// ---------------------------------------------------------------- convert
__global__ void k_cvt_bf16(bf16_t* __restrict__ dst, const float* __restrict__ src,
                           long n, long npad) {
  long i = (long)blockIdx.x * blockDim.x + threadIdx.x;
  long stride = (long)gridDim.x * blockDim.x;
  for (; i < npad; i += stride)
    dst[i] = (i < n) ? (bf16_t)src[i] : (bf16_t)0.0f;
}

// src[R][C] f32  ->  dst[C][R] bf16  (K-contiguous weight layout)
__global__ void k_cvt_T(bf16_t* __restrict__ dst, const float* __restrict__ src,
                        int R, int C) {
  long total = (long)R * C;
  long i = (long)blockIdx.x * blockDim.x + threadIdx.x;
  long stride = (long)gridDim.x * blockDim.x;
  for (; i < total; i += stride) {
    int r = (int)(i / C), c = (int)(i % C);
    dst[(long)c * R + r] = (bf16_t)src[i];
  }
}

__global__ void k_head_bias(float* __restrict__ dst, const float* __restrict__ b0,
                            const float* __restrict__ cb) {
  int v = blockIdx.x * blockDim.x + threadIdx.x;
  if (v < VHEADP)
    dst[v] = (v < 20000) ? b0[v] : ((v < VHEAD) ? cb[v - 20000] : 0.0f);
}

// ---------------------------------------------------------------- async copy helpers
// CDNA5 async global->LDS DMA: 16B per lane, tracked by ASYNCcnt (ISA 15.18.3 op 98).
// LDS operand = low 32 bits of the generic pointer (flat->LDS aperture keeps the
// byte offset in addr[31:0], ISA 10.2).
__device__ inline void async_ld_b128(const bf16_t* lds_generic, const bf16_t* g) {
  unsigned lds = (unsigned)(uintptr_t)lds_generic;
  unsigned long long ga = (unsigned long long)(uintptr_t)g;
  asm volatile("global_load_async_to_lds_b128 %0, %1, off"
               :: "v"(lds), "v"(ga) : "memory");
}
__device__ inline void wait_async0() {
  asm volatile("s_wait_asynccnt 0x0" ::: "memory");
}

__device__ inline v16bf join16(v8bf lo, v8bf hi) {
  union { v16bf v; v8bf h[2]; } u;
  u.h[0] = lo; u.h[1] = hi;
  return u.v;
}

// ---------------------------------------------------------------- GEMM
// C[M,N] = A[M,K](bf16) * W[N,K](bf16)^T
// LOGSUM=false : store bf16 C (cols < Nout)
// LOGSUM=true  : fused stable logsumexp partials over cols < Nout(=V), bias added
template <bool LOGSUM>
__global__ void __launch_bounds__(256)
k_gemm(const bf16_t* __restrict__ A, const bf16_t* __restrict__ W, int K,
       int Nout, const float* __restrict__ bias, bf16_t* __restrict__ outbf,
       float* __restrict__ pmax, float* __restrict__ psum, int TN) {
  constexpr int BM = 128, BN = 128, BK = 32, BKP = 40;  // +8 halfs pad: conflict-free b128
  __shared__ bf16_t sA[2][BM * BKP];
  __shared__ bf16_t sW[2][BN * BKP];
  __shared__ float  sred[BM][2][2];

  const int tid   = threadIdx.x;
  const int lane  = tid & 31;
  const int wid   = tid >> 5;
  const int waveM = wid >> 1;      // 0..3  (rows of 32)
  const int waveN = wid & 1;       // 0..1  (cols of 64)
  const int hlf   = lane >> 4;     // 0/1
  const int l15   = lane & 15;
  const int bm = blockIdx.y, bn = blockIdx.x;
  const long arow0 = (long)bm * BM;
  const long wrow0 = (long)bn * BN;

  // per-thread staging chunks: c in [0,512): row = c>>2, 8-half chunk (c&3)*8
  const int c0row = tid >> 2,          c0off = (tid & 3) * 8;
  const int c1row = (tid + 256) >> 2,  c1off = ((tid + 256) & 3) * 8;
  const bf16_t* Ag0 = A + (arow0 + c0row) * K + c0off;
  const bf16_t* Ag1 = A + (arow0 + c1row) * K + c1off;
  const bf16_t* Wg0 = W + (wrow0 + c0row) * K + c0off;
  const bf16_t* Wg1 = W + (wrow0 + c1row) * K + c1off;
  const int l0 = c0row * BKP + c0off;
  const int l1 = c1row * BKP + c1off;

  v8f acc[2][4];
#pragma unroll
  for (int mi = 0; mi < 2; ++mi)
#pragma unroll
    for (int ni = 0; ni < 4; ++ni) acc[mi][ni] = (v8f)0.0f;

  const int KS = K >> 5;

  // stage tile 0
  async_ld_b128(&sA[0][l0], Ag0);
  async_ld_b128(&sA[0][l1], Ag1);
  async_ld_b128(&sW[0][l0], Wg0);
  async_ld_b128(&sW[0][l1], Wg1);

  for (int ks = 0; ks < KS; ++ks) {
    const int buf = ks & 1;
    wait_async0();      // own stage-ks transfers landed in LDS
    __syncthreads();    // all waves' stage-ks landed; all compute(ks-1) finished

    if (ks + 1 < KS) {  // async prefetch of next tile overlaps the WMMAs below
      int k0 = (ks + 1) * BK;
      async_ld_b128(&sA[buf ^ 1][l0], Ag0 + k0);
      async_ld_b128(&sA[buf ^ 1][l1], Ag1 + k0);
      async_ld_b128(&sW[buf ^ 1][l0], Wg0 + k0);
      async_ld_b128(&sW[buf ^ 1][l1], Wg1 + k0);
    }

    // fragments from LDS
    v16bf af[2], bfm[4];
#pragma unroll
    for (int mi = 0; mi < 2; ++mi) {
      int r = waveM * 32 + mi * 16 + l15;
      int base = r * BKP + hlf * 8;
      af[mi] = join16(*(const v8bf*)(&sA[buf][base]),
                      *(const v8bf*)(&sA[buf][base + 16]));
    }
#pragma unroll
    for (int ni = 0; ni < 4; ++ni) {
      int r = waveN * 64 + ni * 16 + l15;
      int base = r * BKP + hlf * 8;
      bfm[ni] = join16(*(const v8bf*)(&sW[buf][base]),
                       *(const v8bf*)(&sW[buf][base + 16]));
    }
#pragma unroll
    for (int mi = 0; mi < 2; ++mi)
#pragma unroll
      for (int ni = 0; ni < 4; ++ni)
        acc[mi][ni] = __builtin_amdgcn_wmma_f32_16x16x32_bf16(
            false, af[mi], false, bfm[ni], (short)0, acc[mi][ni], false, false);
  }

  if (!LOGSUM) {
#pragma unroll
    for (int mi = 0; mi < 2; ++mi)
#pragma unroll
      for (int ni = 0; ni < 4; ++ni) {
        int col = bn * BN + waveN * 64 + ni * 16 + l15;
        if (col < Nout) {
#pragma unroll
          for (int r = 0; r < 8; ++r) {
            long row = arow0 + waveM * 32 + mi * 16 + hlf * 8 + r;
            outbf[row * Nout + col] = (bf16_t)acc[mi][ni][r];
          }
        }
      }
    return;
  }

  // ---- fused stable logsumexp over this block's 128 columns ----
  float rm[2][8], rs[2][8];
#pragma unroll
  for (int mi = 0; mi < 2; ++mi) {
#pragma unroll
    for (int r = 0; r < 8; ++r) rm[mi][r] = -3.4e38f;
#pragma unroll
    for (int ni = 0; ni < 4; ++ni) {
      int col = bn * BN + waveN * 64 + ni * 16 + l15;
      bool valid = col < Nout;
      float bv = valid ? bias[col] : 0.0f;
#pragma unroll
      for (int r = 0; r < 8; ++r) {
        float x = valid ? (acc[mi][ni][r] + bv) : -3.4e38f;
        acc[mi][ni][r] = x;
        rm[mi][r] = fmaxf(rm[mi][r], x);
      }
    }
#pragma unroll
    for (int r = 0; r < 8; ++r) {   // row max across the 16 lanes of each half
      rm[mi][r] = fmaxf(rm[mi][r], __shfl_xor(rm[mi][r], 1, 32));
      rm[mi][r] = fmaxf(rm[mi][r], __shfl_xor(rm[mi][r], 2, 32));
      rm[mi][r] = fmaxf(rm[mi][r], __shfl_xor(rm[mi][r], 4, 32));
      rm[mi][r] = fmaxf(rm[mi][r], __shfl_xor(rm[mi][r], 8, 32));
      rs[mi][r] = 0.0f;
    }
#pragma unroll
    for (int ni = 0; ni < 4; ++ni) {
      int col = bn * BN + waveN * 64 + ni * 16 + l15;
      bool valid = col < Nout;
#pragma unroll
      for (int r = 0; r < 8; ++r)
        if (valid) rs[mi][r] += __expf(acc[mi][ni][r] - rm[mi][r]);
    }
#pragma unroll
    for (int r = 0; r < 8; ++r) {
      rs[mi][r] += __shfl_xor(rs[mi][r], 1, 32);
      rs[mi][r] += __shfl_xor(rs[mi][r], 2, 32);
      rs[mi][r] += __shfl_xor(rs[mi][r], 4, 32);
      rs[mi][r] += __shfl_xor(rs[mi][r], 8, 32);
    }
  }
  if (l15 == 0) {
#pragma unroll
    for (int mi = 0; mi < 2; ++mi)
#pragma unroll
      for (int r = 0; r < 8; ++r) {
        int rl = waveM * 32 + mi * 16 + hlf * 8 + r;
        sred[rl][waveN][0] = rm[mi][r];
        sred[rl][waveN][1] = rs[mi][r];
      }
  }
  __syncthreads();
  if (tid < BM) {
    float m0 = sred[tid][0][0], s0 = sred[tid][0][1];
    float m1 = sred[tid][1][0], s1 = sred[tid][1][1];
    float M = fmaxf(m0, m1);
    float S = s0 * __expf(m0 - M) + s1 * __expf(m1 - M);
    long row = arow0 + tid;
    pmax[row * TN + bn] = M;
    psum[row * TN + bn] = S;
  }
}

// ---------------------------------------------------------------- gather picked logits
__global__ void __launch_bounds__(128)
k_gather(const bf16_t* __restrict__ hp0, const bf16_t* __restrict__ hp1,
         const bf16_t* __restrict__ hp2, const bf16_t* __restrict__ headW,
         const bf16_t* __restrict__ w1, const bf16_t* __restrict__ w2,
         const float* __restrict__ bh, const float* __restrict__ b1,
         const float* __restrict__ b2, const int* __restrict__ target,
         float* __restrict__ pickA, float* __restrict__ pickB) {
  __shared__ float red[128];
  const int n = blockIdx.x, tid = threadIdx.x;
  const int t = target[n];
  int vA, tail, vB;
  if (t < 20000)      { vA = t;     tail = 0; vB = 0; }
  else if (t < 50000) { vA = 20001; tail = 1; vB = t - 20000; }  // head_lp[:, -1]
  else                { vA = 20000; tail = 2; vB = t - 50000; }  // head_lp[:, -2]

  float p = 0.0f;
  {
    const bf16_t* a = hp0 + (long)n * DIM;
    const bf16_t* w = headW + (long)vA * DIM;
    for (int j = tid; j < DIM; j += 128) p += (float)a[j] * (float)w[j];
  }
  red[tid] = p; __syncthreads();
  for (int s = 64; s > 0; s >>= 1) { if (tid < s) red[tid] += red[tid + s]; __syncthreads(); }
  if (tid == 0) pickA[n] = red[0] + bh[vA];
  __syncthreads();

  float q = 0.0f;
  if (tail == 1) {
    const bf16_t* a = hp1 + (long)n * 256;
    const bf16_t* w = w1 + (long)vB * 256;
    for (int j = tid; j < 256; j += 128) q += (float)a[j] * (float)w[j];
  } else if (tail == 2) {
    const bf16_t* a = hp2 + (long)n * 64;
    const bf16_t* w = w2 + (long)vB * 64;
    for (int j = tid; j < 64; j += 128) q += (float)a[j] * (float)w[j];
  }
  red[tid] = q; __syncthreads();
  for (int s = 64; s > 0; s >>= 1) { if (tid < s) red[tid] += red[tid + s]; __syncthreads(); }
  if (tid == 0)
    pickB[n] = red[0] + (tail == 1 ? b1[vB] : (tail == 2 ? b2[vB] : 0.0f));
}

// ---------------------------------------------------------------- final combine (1 wave / token)
__global__ void __launch_bounds__(32)
k_final(const float* __restrict__ pmH, const float* __restrict__ psH,
        const float* __restrict__ pm1, const float* __restrict__ ps1,
        const float* __restrict__ pm2, const float* __restrict__ ps2,
        const float* __restrict__ pickA, const float* __restrict__ pickB,
        const int* __restrict__ target, float* __restrict__ out) {
  const int n = blockIdx.x, lane = threadIdx.x;
  const int t = target[n];

  auto lse = [&](const float* pm, const float* ps, int TN) -> float {
    float m = -3.4e38f;
    for (int i = lane; i < TN; i += 32) m = fmaxf(m, pm[(long)n * TN + i]);
    for (int k = 16; k >= 1; k >>= 1) m = fmaxf(m, __shfl_xor(m, k, 32));
    float s = 0.0f;
    for (int i = lane; i < TN; i += 32)
      s += ps[(long)n * TN + i] * __expf(pm[(long)n * TN + i] - m);
    for (int k = 16; k >= 1; k >>= 1) s += __shfl_xor(s, k, 32);
    return m + __logf(s);
  };

  float lseH = lse(pmH, psH, TNH);
  float r;
  if (t < 20000) {
    r = -(pickA[n] - lseH);
  } else if (t < 50000) {
    float lseT = lse(pm1, ps1, TN1);
    r = -((pickA[n] - lseH) + (pickB[n] - lseT));
  } else {
    float lseT = lse(pm2, ps2, TN2);
    r = -((pickA[n] - lseH) + (pickB[n] - lseT));
  }
  if (lane == 0) out[n] = r;
}

// ---------------------------------------------------------------- launcher
extern "C" void kernel_launch(void* const* d_in, const int* in_sizes, int n_in,
                              void* d_out, int out_size, void* d_ws, size_t ws_size,
                              hipStream_t stream) {
  (void)in_sizes; (void)n_in; (void)out_size; (void)ws_size;
  const float* hidden = (const float*)d_in[0];
  const int*   target = (const int*)d_in[1];
  const float* w0 = (const float*)d_in[2];
  const float* b0 = (const float*)d_in[3];
  const float* cw = (const float*)d_in[4];
  const float* cb = (const float*)d_in[5];
  const float* p0 = (const float*)d_in[6];
  const float* w1 = (const float*)d_in[7];
  const float* b1 = (const float*)d_in[8];
  const float* p1 = (const float*)d_in[9];
  const float* w2 = (const float*)d_in[10];
  const float* b2 = (const float*)d_in[11];
  const float* p2 = (const float*)d_in[12];
  float* out = (float*)d_out;

  char* ws = (char*)d_ws;
  size_t off = 0;
  auto alloc = [&](size_t bytes) -> char* {
    char* p = ws + off;
    off = (off + bytes + 255) & ~((size_t)255);
    return p;
  };
  bf16_t* Hbf   = (bf16_t*)alloc((size_t)NTOK * DIM * 2);
  bf16_t* hp0   = (bf16_t*)alloc((size_t)NTOK * DIM * 2);
  bf16_t* hp1   = (bf16_t*)alloc((size_t)NTOK * 256 * 2);
  bf16_t* hp2   = (bf16_t*)alloc((size_t)NTOK * 64 * 2);
  bf16_t* headW = (bf16_t*)alloc((size_t)VHEADP * DIM * 2);
  bf16_t* p0T   = (bf16_t*)alloc((size_t)1024 * DIM * 2);
  bf16_t* p1T   = (bf16_t*)alloc((size_t)256 * DIM * 2);
  bf16_t* p2T   = (bf16_t*)alloc((size_t)128 * DIM * 2);   // 64 real + 64 pad rows
  bf16_t* w1b   = (bf16_t*)alloc((size_t)V1P * 256 * 2);
  bf16_t* w2b   = (bf16_t*)alloc((size_t)V2P * 64 * 2);
  float*  bh    = (float*)alloc((size_t)VHEADP * 4);
  float*  pmH   = (float*)alloc((size_t)NTOK * TNH * 4);
  float*  psH   = (float*)alloc((size_t)NTOK * TNH * 4);
  float*  pm1   = (float*)alloc((size_t)NTOK * TN1 * 4);
  float*  ps1   = (float*)alloc((size_t)NTOK * TN1 * 4);
  float*  pm2   = (float*)alloc((size_t)NTOK * TN2 * 4);
  float*  ps2   = (float*)alloc((size_t)NTOK * TN2 * 4);
  float*  pickA = (float*)alloc((size_t)NTOK * 4);
  float*  pickB = (float*)alloc((size_t)NTOK * 4);

  const int CT = 256, CB = 2048;
  // bf16 conversions (zero-padded vocab rows)
  k_cvt_bf16<<<CB, CT, 0, stream>>>(Hbf, hidden, (long)NTOK * DIM, (long)NTOK * DIM);
  k_cvt_bf16<<<CB, CT, 0, stream>>>(headW, w0, (long)20000 * DIM, (long)20000 * DIM);
  k_cvt_bf16<<<8, CT, 0, stream>>>(headW + (long)20000 * DIM, cw, 2 * DIM, 2 * DIM);
  k_cvt_bf16<<<512, CT, 0, stream>>>(headW + (long)VHEAD * DIM, (const float*)nullptr,
                                     0, (long)(VHEADP - VHEAD) * DIM);
  k_cvt_T<<<CB, CT, 0, stream>>>(p0T, p0, 1024, 1024);
  k_cvt_T<<<CB, CT, 0, stream>>>(p1T, p1, 1024, 256);
  k_cvt_T<<<512, CT, 0, stream>>>(p2T, p2, 1024, 64);
  k_cvt_bf16<<<512, CT, 0, stream>>>(p2T + (long)64 * DIM, (const float*)nullptr,
                                     0, (long)64 * DIM);
  k_cvt_bf16<<<CB, CT, 0, stream>>>(w1b, w1, (long)V1 * 256, (long)V1P * 256);
  k_cvt_bf16<<<CB, CT, 0, stream>>>(w2b, w2, (long)V2 * 64, (long)V2P * 64);
  k_head_bias<<<(VHEADP + CT - 1) / CT, CT, 0, stream>>>(bh, b0, cb);

  // projection GEMMs: hp = H * projT^T
  k_gemm<false><<<dim3(8, 16), 256, 0, stream>>>(Hbf, p0T, DIM, 1024, nullptr, hp0,
                                                 nullptr, nullptr, 0);
  k_gemm<false><<<dim3(2, 16), 256, 0, stream>>>(Hbf, p1T, DIM, 256, nullptr, hp1,
                                                 nullptr, nullptr, 0);
  k_gemm<false><<<dim3(1, 16), 256, 0, stream>>>(Hbf, p2T, DIM, 64, nullptr, hp2,
                                                 nullptr, nullptr, 0);
  // logit GEMMs with fused logsumexp partials
  k_gemm<true><<<dim3(TNH, 16), 256, 0, stream>>>(hp0, headW, DIM, VHEAD, bh, nullptr,
                                                  pmH, psH, TNH);
  k_gemm<true><<<dim3(TN1, 16), 256, 0, stream>>>(hp1, w1b, 256, V1, b1, nullptr,
                                                  pm1, ps1, TN1);
  k_gemm<true><<<dim3(TN2, 16), 256, 0, stream>>>(hp2, w2b, 64, V2, b2, nullptr,
                                                  pm2, ps2, TN2);
  // picked logits + final nll
  k_gather<<<NTOK, 128, 0, stream>>>(hp0, hp1, hp2, headW, w1b, w2b, bh, b1, b2,
                                     target, pickA, pickB);
  k_final<<<NTOK, 32, 0, stream>>>(pmH, psH, pm1, ps1, pm2, ps2, pickA, pickB,
                                   target, out);
}
